// SoftDTWAlignment_31353261261228
// MI455X (gfx1250) — compile-verified, matrix operand order checked by reading, not run on hardware
//
#include <hip/hip_runtime.h>
#include <hip/hip_bf16.h>

#define M_DIM 4096
#define N_DIM 4096
#define D_DIM 512
#define BIGF 1e9f

typedef __attribute__((ext_vector_type(16))) _Float16 v16h;
typedef __attribute__((ext_vector_type(8)))  _Float16 v8h;
typedef __attribute__((ext_vector_type(8)))  float    v8f;

// ---------------------------------------------------------------------------
// Kernel 1: row L2-normalize + split fp32 -> (f16 hi, f16 lo)  (Markidis)
// One 256-thread block per row (8192 rows total: x rows then y rows).
// ---------------------------------------------------------------------------
__global__ __launch_bounds__(256) void normalize_split_kernel(
    const float* __restrict__ x, const float* __restrict__ y,
    _Float16* __restrict__ xh, _Float16* __restrict__ xl,
    _Float16* __restrict__ yh, _Float16* __restrict__ yl) {
  __shared__ float red[256];
  const int row = blockIdx.x;
  const int t = threadIdx.x;

  const float* src;
  _Float16 *oh, *ol;
  if (row < M_DIM) {
    src = x + (size_t)row * D_DIM;
    oh = xh + (size_t)row * D_DIM;
    ol = xl + (size_t)row * D_DIM;
  } else {
    const int r = row - M_DIM;
    src = y + (size_t)r * D_DIM;
    oh = yh + (size_t)r * D_DIM;
    ol = yl + (size_t)r * D_DIM;
  }

  const float v0 = src[t];
  const float v1 = src[t + 256];
  red[t] = v0 * v0 + v1 * v1;
  __syncthreads();
  #pragma unroll
  for (int s = 128; s > 0; s >>= 1) {
    if (t < s) red[t] += red[t + s];
    __syncthreads();
  }
  const float inv = 1.0f / fmaxf(sqrtf(red[0]), 1e-12f);

  {
    const float v = v0 * inv;
    const _Float16 h = (_Float16)v;
    oh[t] = h;
    ol[t] = (_Float16)(v - (float)h);
  }
  {
    const float v = v1 * inv;
    const _Float16 h = (_Float16)v;
    oh[t + 256] = h;
    ol[t + 256] = (_Float16)(v - (float)h);
  }
}

// ---------------------------------------------------------------------------
// Fragment loader: 16-bit A/B 16x32 WMMA layout (wave32).
// Lanes 0-15 (row = lane&15): K = k0+{0..7, 16..23}
// Lanes 16-31:                K = k0+{8..15, 24..31}
// Two global_load_b128 per fragment.
// ---------------------------------------------------------------------------
__device__ __forceinline__ v16h load_frag(const _Float16* __restrict__ base,
                                          int row, int k0, int lane) {
  const _Float16* p = base + (size_t)row * D_DIM + k0 + ((lane >> 4) << 3);
  const v8h a0 = *reinterpret_cast<const v8h*>(p);
  const v8h a1 = *reinterpret_cast<const v8h*>(p + 16);
  v16h r;
  #pragma unroll
  for (int e = 0; e < 8; ++e) { r[e] = a0[e]; r[e + 8] = a1[e]; }
  return r;
}

// ---------------------------------------------------------------------------
// Kernel 2: cost = 1 - xn . yn^T via split-f16 WMMA (hi*hi + hi*lo + lo*hi).
// Block = 256 threads = 8 waves (2x4); each wave owns a 32x32 output tile
// (2x2 accumulators of 16x16). Block tile = 64 x 128.
// ---------------------------------------------------------------------------
__global__ __launch_bounds__(256) void cost_gemm_kernel(
    const _Float16* __restrict__ xh, const _Float16* __restrict__ xl,
    const _Float16* __restrict__ yh, const _Float16* __restrict__ yl,
    float* __restrict__ cost) {
  const int lane = threadIdx.x & 31;
  const int wave = threadIdx.x >> 5;          // 0..7
  const int wr = wave >> 2;                   // 0..1
  const int wc = wave & 3;                    // 0..3
  const int m0 = blockIdx.y * 64 + wr * 32;
  const int n0 = blockIdx.x * 128 + wc * 32;
  const int r16 = lane & 15;

  v8f acc[2][2];
  #pragma unroll
  for (int i = 0; i < 2; ++i)
    #pragma unroll
    for (int j = 0; j < 2; ++j) acc[i][j] = (v8f){0,0,0,0,0,0,0,0};

  for (int k0 = 0; k0 < D_DIM; k0 += 32) {
    v16h ah[2], al[2], bh[2], bl[2];
    #pragma unroll
    for (int mt = 0; mt < 2; ++mt) {
      ah[mt] = load_frag(xh, m0 + mt * 16 + r16, k0, lane);
      al[mt] = load_frag(xl, m0 + mt * 16 + r16, k0, lane);
    }
    #pragma unroll
    for (int nt = 0; nt < 2; ++nt) {
      bh[nt] = load_frag(yh, n0 + nt * 16 + r16, k0, lane);
      bl[nt] = load_frag(yl, n0 + nt * 16 + r16, k0, lane);
    }
    #pragma unroll
    for (int mt = 0; mt < 2; ++mt)
      #pragma unroll
      for (int nt = 0; nt < 2; ++nt) {
        acc[mt][nt] = __builtin_amdgcn_wmma_f32_16x16x32_f16(
            false, ah[mt], false, bh[nt], (short)0, acc[mt][nt], false, false);
        acc[mt][nt] = __builtin_amdgcn_wmma_f32_16x16x32_f16(
            false, ah[mt], false, bl[nt], (short)0, acc[mt][nt], false, false);
        acc[mt][nt] = __builtin_amdgcn_wmma_f32_16x16x32_f16(
            false, al[mt], false, bh[nt], (short)0, acc[mt][nt], false, false);
      }
  }

  // C/D 16x16 f32 layout: VGPR r -> row r (lanes 0-15) / row r+8 (lanes 16-31),
  // column = lane & 15.
  #pragma unroll
  for (int mt = 0; mt < 2; ++mt)
    #pragma unroll
    for (int nt = 0; nt < 2; ++nt) {
      const int col = n0 + nt * 16 + r16;
      const int rowBase = m0 + mt * 16 + ((lane >> 4) << 3);
      #pragma unroll
      for (int r = 0; r < 8; ++r)
        cost[(size_t)(rowBase + r) * N_DIM + col] = 1.0f - acc[mt][nt][r];
    }
}

// ---------------------------------------------------------------------------
// Kernel 3: forward soft-DTW DP, anti-diagonal wavefront.
// Single 1024-thread workgroup; thread t owns 1-based columns 4t+1..4t+4.
// Keeps diagonals d-1 (p1) and d-2 (p2) in registers; boundary column via LDS.
// ---------------------------------------------------------------------------
__global__ __launch_bounds__(1024) void dp_forward_kernel(
    const float* __restrict__ cost, float* __restrict__ dpf,
    float* __restrict__ scal, const float* __restrict__ gamma) {
  __shared__ float l1[1024], l2[1024];
  const int t = threadIdx.x;
  const float g = fmaxf(fabsf(gamma[0]), 1e-4f);
  const float ginv = 1.0f / g;
  const int jbase = 4 * t + 1;

  float p1[4], p2[4];
  #pragma unroll
  for (int c = 0; c < 4; ++c) { p1[c] = BIGF; p2[c] = BIGF; }

  for (int d = 2; d <= M_DIM + N_DIM; ++d) {
    l1[t] = p1[3];
    l2[t] = p2[3];
    __syncthreads();
    const float left1 = (t == 0) ? BIGF : l1[t - 1];
    const float left2 = (t == 0) ? BIGF : l2[t - 1];
    float nw[4];
    #pragma unroll
    for (int c = 0; c < 4; ++c) {
      const int j = jbase + c;
      const int i = d - j;
      float rl = (c == 0) ? left1 : p1[c - 1];   // R[i, j-1]   (diag d-1)
      float rd = (c == 0) ? left2 : p2[c - 1];   // R[i-1, j-1] (diag d-2)
      if (i == 1 && j == 1) rd = 0.0f;           // R[0,0] = 0
      const float ru = p1[c];                    // R[i-1, j]   (diag d-1)
      float val = BIGF;
      if (i >= 1 && i <= M_DIM) {
        const size_t idx = (size_t)(i - 1) * N_DIM + (j - 1);
        const float cst = cost[idx];
        if (i < M_DIM) __builtin_prefetch(&cost[idx + N_DIM], 0, 0);
        const float mn = fminf(rd, fminf(ru, rl));
        const float s = __expf((mn - rd) * ginv) + __expf((mn - ru) * ginv) +
                        __expf((mn - rl) * ginv);
        val = cst + mn - g * __logf(s);
        dpf[idx] = val;
      }
      nw[c] = val;
    }
    #pragma unroll
    for (int c = 0; c < 4; ++c) { p2[c] = p1[c]; p1[c] = nw[c]; }
    __syncthreads();
  }
  if (t == 1023) scal[0] = p1[3];  // distance = R[M, N]
}

// ---------------------------------------------------------------------------
// Kernel 4: backward DP fused with alignment emission + sum reduction.
// E[i,j] = softmin(E[i+1,j+1]+c(i+1,j+1), E[i+1,j]+c(i+1,j), E[i,j+1]+c(i,j+1))
// Emits a = exp(min(-(dpf + cost + E - dist)/g, 0)) directly to d_out.
// ---------------------------------------------------------------------------
__global__ __launch_bounds__(1024) void dp_backward_emit_kernel(
    const float* __restrict__ cost, const float* __restrict__ dpf,
    float* __restrict__ scal, const float* __restrict__ gamma,
    float* __restrict__ aln) {
  __shared__ float l1[1024], l2[1024];
  __shared__ float red[1024];
  const int t = threadIdx.x;
  const float g = fmaxf(fabsf(gamma[0]), 1e-4f);
  const float ginv = 1.0f / g;
  const float dist = scal[0];
  const int jbase = 4 * t + 1;

  float n1[4], n2[4];
  #pragma unroll
  for (int c = 0; c < 4; ++c) { n1[c] = BIGF; n2[c] = BIGF; }
  float acc = 0.0f;
  if (t == 1023) {
    n1[3] = 0.0f;  // E[M,N] = 0 on diagonal d = M+N
    const size_t idx = (size_t)(M_DIM - 1) * N_DIM + (N_DIM - 1);
    const float a =
        __expf(fminf(-(dpf[idx] + cost[idx] + 0.0f - dist) * ginv, 0.0f));
    aln[idx] = a;
    acc += a;
  }

  for (int d = M_DIM + N_DIM - 1; d >= 2; --d) {
    l1[t] = n1[0];
    l2[t] = n2[0];
    __syncthreads();
    const float r1 = (t == 1023) ? BIGF : l1[t + 1];
    const float r2 = (t == 1023) ? BIGF : l2[t + 1];
    float nw[4];
    #pragma unroll
    for (int c = 0; c < 4; ++c) {
      const int j = jbase + c;
      const int i = d - j;
      const float e_dr = (c == 3) ? r2 : n2[c + 1];  // E[i+1, j+1]
      const float e_dn = n1[c];                      // E[i+1, j]
      const float e_rt = (c == 3) ? r1 : n1[c + 1];  // E[i, j+1]
      float val = BIGF;
      if (i >= 1 && i <= M_DIM) {
        const float c1 = (i < M_DIM && j < N_DIM)
                             ? e_dr + cost[(size_t)i * N_DIM + j] : BIGF;
        const float c2 = (i < M_DIM)
                             ? e_dn + cost[(size_t)i * N_DIM + (j - 1)] : BIGF;
        const float c3 = (j < N_DIM)
                             ? e_rt + cost[(size_t)(i - 1) * N_DIM + j] : BIGF;
        const float mn = fminf(c1, fminf(c2, c3));
        const float s = __expf((mn - c1) * ginv) + __expf((mn - c2) * ginv) +
                        __expf((mn - c3) * ginv);
        val = mn - g * __logf(s);
        const size_t idx = (size_t)(i - 1) * N_DIM + (j - 1);
        const float a =
            __expf(fminf(-(dpf[idx] + cost[idx] + val - dist) * ginv, 0.0f));
        aln[idx] = a;
        acc += a;
      }
      nw[c] = val;
    }
    #pragma unroll
    for (int c = 0; c < 4; ++c) { n2[c] = n1[c]; n1[c] = nw[c]; }
    __syncthreads();
  }

  red[t] = acc;
  __syncthreads();
  #pragma unroll
  for (int s = 512; s > 0; s >>= 1) {
    if (t < s) red[t] += red[t + s];
    __syncthreads();
  }
  if (t == 0) scal[1] = red[0];  // total
}

// ---------------------------------------------------------------------------
// Kernel 5: normalize alignment in place, append distance scalar.
// ---------------------------------------------------------------------------
__global__ __launch_bounds__(256) void finalize_kernel(
    float* __restrict__ aln, const float* __restrict__ scal) {
  const float scale = (float)M_DIM / fmaxf(scal[1], 1e-8f);
  const size_t total = (size_t)M_DIM * N_DIM;
  const size_t stride = (size_t)gridDim.x * blockDim.x;
  for (size_t k = (size_t)blockIdx.x * blockDim.x + threadIdx.x; k < total;
       k += stride)
    aln[k] *= scale;
  if (blockIdx.x == 0 && threadIdx.x == 0) aln[total] = scal[0];
}

// ---------------------------------------------------------------------------
extern "C" void kernel_launch(void* const* d_in, const int* in_sizes, int n_in,
                              void* d_out, int out_size, void* d_ws,
                              size_t ws_size, hipStream_t stream) {
  (void)in_sizes; (void)n_in; (void)out_size; (void)ws_size;
  const float* x = (const float*)d_in[0];
  const float* y = (const float*)d_in[1];
  const float* gamma = (const float*)d_in[2];
  float* out = (float*)d_out;

  char* w = (char*)d_ws;
  const size_t half_mat = (size_t)M_DIM * D_DIM * sizeof(_Float16);  // 4 MB
  _Float16* xh = (_Float16*)w;               w += half_mat;
  _Float16* xl = (_Float16*)w;               w += half_mat;
  _Float16* yh = (_Float16*)w;               w += half_mat;
  _Float16* yl = (_Float16*)w;               w += half_mat;
  float* cost = (float*)w;                   w += (size_t)M_DIM * N_DIM * 4;
  float* dpf  = (float*)w;                   w += (size_t)M_DIM * N_DIM * 4;
  float* scal = (float*)w;                   // [0]=distance, [1]=total

  normalize_split_kernel<<<M_DIM + N_DIM, 256, 0, stream>>>(x, y, xh, xl, yh, yl);
  cost_gemm_kernel<<<dim3(N_DIM / 128, M_DIM / 64), 256, 0, stream>>>(
      xh, xl, yh, yl, cost);
  dp_forward_kernel<<<1, 1024, 0, stream>>>(cost, dpf, scal, gamma);
  dp_backward_emit_kernel<<<1, 1024, 0, stream>>>(cost, dpf, scal, gamma, out);
  finalize_kernel<<<4096, 256, 0, stream>>>(out, scal);
}
